// ClusterFeatureExtractor_10634339025515
// MI455X (gfx1250) — compile-verified, matrix operand order checked by reading, not run on hardware
//
#include <hip/hip_runtime.h>
#include <hip/hip_bf16.h>

// ---------------------------------------------------------------- constants
#define B_    16
#define N_    2048
#define D_    1024
#define H_    16
#define DH_   64
#define C_    5
#define MAXT_ 1000
#define S_    512
#define DFF_  4096
#define CB_   (C_ * B_)      // 80
#define M_    (CB_ * S_)     // 40960
#define LN_EPS 1e-12f

typedef __attribute__((ext_vector_type(16))) __bf16 v16bf;
typedef __attribute__((ext_vector_type(8)))  __bf16 v8bf;
typedef __attribute__((ext_vector_type(8)))  float  v8f;
typedef __attribute__((ext_vector_type(4)))  __bf16 v4bf;

// d_out layout (floats): cluster | cluster_mask | pooled
#define OFF_MASK  ((size_t)C_ * B_ * MAXT_ * D_)          // 81,920,000
#define OFF_POOL  (OFF_MASK + (size_t)C_ * B_ * MAXT_)    // 82,000,000

// LDS byte address of a shared-memory pointer: flat LDS aperture keeps the
// LDS offset in addr[31:0] (ISA 10.2 aperture table), so truncation suffices.
__device__ __forceinline__ unsigned lds_addr_u32(const void* p) {
  return (unsigned)(unsigned long long)p;
}

// async global->LDS 32B (two b128, offset applies to BOTH global and LDS addr)
__device__ __forceinline__ void async_copy_32B(unsigned lds, const void* gptr) {
  asm volatile("global_load_async_to_lds_b128 %0, %1, off\n\t"
               "global_load_async_to_lds_b128 %0, %1, off offset:16"
               :: "v"(lds), "v"(gptr) : "memory");
}

__device__ __forceinline__ void wait_asynccnt0() {
#if __has_builtin(__builtin_amdgcn_s_wait_asynccnt)
  __builtin_amdgcn_s_wait_asynccnt(0);
#else
  asm volatile("s_wait_asynccnt 0x0" ::: "memory");
#endif
}

// ---------------------------------------------------------------- kernel 1: gather + build X (bf16)
__global__ __launch_bounds__(256)
void gather_kernel(const float* __restrict__ vision, const int* __restrict__ cstate,
                   float* __restrict__ cluster_out, float* __restrict__ mask_out,
                   __bf16* __restrict__ Xb) {
  __shared__ int s_idx[MAXT_];
  __shared__ int s_cnt;
  const int cb  = blockIdx.x;
  const int c   = cb / B_;
  const int b   = cb % B_;
  const int tid = threadIdx.x;

  if (tid == 0) {
    int cnt = 0;
    const int* row = cstate + (size_t)b * N_;
    for (int n = 0; n < N_; ++n) {
      if (row[n] == c) {
        if (cnt < MAXT_) s_idx[cnt] = n;
        ++cnt;
      }
    }
    s_cnt = cnt < MAXT_ ? cnt : MAXT_;
  }
  __syncthreads();
  const int cnt = s_cnt;

  // cluster_mask
  for (int t = tid; t < MAXT_; t += 256)
    mask_out[(size_t)cb * MAXT_ + t] = (t < cnt) ? 1.0f : 0.0f;

  // X row 0 = zeros (prefix). 256 threads x 4 bf16 = 1024
  {
    v4bf z;
    z[0] = (__bf16)0.0f; z[1] = (__bf16)0.0f; z[2] = (__bf16)0.0f; z[3] = (__bf16)0.0f;
    *(v4bf*)(Xb + (size_t)cb * S_ * D_ + (size_t)tid * 4) = z;
  }

  const float4* src4 = (const float4*)(vision + (size_t)b * N_ * D_);
  float4* dst4 = (float4*)(cluster_out + (size_t)cb * MAXT_ * D_);
  for (int t = 0; t < MAXT_; ++t) {
    float4 val;
    if (t < cnt) {
      val = src4[(size_t)s_idx[t] * (D_ / 4) + tid];
    } else {
      val = make_float4(0.f, 0.f, 0.f, 0.f);
    }
    dst4[(size_t)t * (D_ / 4) + tid] = val;
    if (t < S_ - 1) {
      v4bf bv;
      bv[0] = (__bf16)val.x; bv[1] = (__bf16)val.y;
      bv[2] = (__bf16)val.z; bv[3] = (__bf16)val.w;
      *(v4bf*)(Xb + (size_t)cb * S_ * D_ + (size_t)(t + 1) * D_ + (size_t)tid * 4) = bv;
    }
  }
}

// ---------------------------------------------------------------- kernel 2: weight convert + transpose to bf16 [N][K]
__global__ __launch_bounds__(256)
void wconv_kernel(const float* __restrict__ W, __bf16* __restrict__ Wt) {
  const int i = blockIdx.x * 256 + threadIdx.x;   // over D_*D_
  const int k = i / D_;
  const int n = i % D_;
  Wt[(size_t)n * D_ + k] = (__bf16)W[i];
}

// ---------------------------------------------------------------- kernel 3: fused K/V WMMA GEMM with async double-buffered LDS
// A: M x 1024 bf16, Btk/Btv: 1024(N) x 1024(K) bf16 (pre-transposed)
#define TILE_K     32
#define LDS_STRIDE 40   // bf16 elems per row; 80 bytes => 16B aligned rows
#define NTILES     (1024 / TILE_K)

__global__ __launch_bounds__(256)
void gemm_kv_fused(const __bf16* __restrict__ A,
                   const __bf16* __restrict__ Btk, const __bf16* __restrict__ Btv,
                   const float* __restrict__ biask, const float* __restrict__ biasv,
                   __bf16* __restrict__ OutK, __bf16* __restrict__ OutV) {
  __shared__ __attribute__((aligned(16))) __bf16 lA[2][128 * LDS_STRIDE];
  __shared__ __attribute__((aligned(16))) __bf16 lK[2][128 * LDS_STRIDE];
  __shared__ __attribute__((aligned(16))) __bf16 lV[2][128 * LDS_STRIDE];

  const int tid   = threadIdx.x;
  const int lane  = tid & 31;
  const int wave  = tid >> 5;
  const int waveM = wave & 3;   // 4 M-groups of 32 rows
  const int waveN = wave >> 2;  // 2 N-groups of 64 cols
  const int g     = lane >> 4;  // half-wave
  const int lm    = lane & 15;
  const int blockN = blockIdx.x * 128;
  const int blockM = blockIdx.y * 128;

  v8f accK[2][4], accV[2][4];
#pragma unroll
  for (int mi = 0; mi < 2; ++mi)
#pragma unroll
    for (int ni = 0; ni < 4; ++ni)
#pragma unroll
      for (int j = 0; j < 8; ++j) { accK[mi][ni][j] = 0.0f; accV[mi][ni][j] = 0.0f; }

  const int ldRow = tid >> 1;          // 0..127
  const int ldCol = (tid & 1) * 16;    // 0 or 16
  const size_t aoff = (size_t)(blockM + ldRow) * 1024 + ldCol;
  const size_t boff = (size_t)(blockN + ldRow) * 1024 + ldCol;
  const unsigned ldsDst = (unsigned)(ldRow * LDS_STRIDE + ldCol) * 2u;

  auto issue_tile = [&](int buf, int k0) {
    async_copy_32B(lds_addr_u32(lA[buf]) + ldsDst, A   + aoff + k0);
    async_copy_32B(lds_addr_u32(lK[buf]) + ldsDst, Btk + boff + k0);
    async_copy_32B(lds_addr_u32(lV[buf]) + ldsDst, Btv + boff + k0);
  };

  issue_tile(0, 0);  // preload tile 0

  for (int i = 0; i < NTILES; ++i) {
    const int cur = i & 1;
    wait_asynccnt0();     // my async writes for tile i have landed in LDS
    __syncthreads();      // everyone's have; also proves compute(i-1) done in all waves
    if (i + 1 < NTILES) issue_tile(cur ^ 1, (i + 1) * TILE_K);

    // fragment loads per ISA 16-bit layout: per lane, K chunks [8g..8g+7], [16+8g..]
    v16bf afrag[2];
#pragma unroll
    for (int mi = 0; mi < 2; ++mi) {
      const __bf16* p = &lA[cur][(waveM * 32 + mi * 16 + lm) * LDS_STRIDE + 8 * g];
      v8bf lo = *(const v8bf*)p;
      v8bf hi = *(const v8bf*)(p + 16);
      afrag[mi] = __builtin_shufflevector(lo, hi, 0,1,2,3,4,5,6,7,8,9,10,11,12,13,14,15);
    }
    v16bf bfrag[4];
#pragma unroll
    for (int ni = 0; ni < 4; ++ni) {
      const __bf16* p = &lK[cur][(waveN * 64 + ni * 16 + lm) * LDS_STRIDE + 8 * g];
      v8bf lo = *(const v8bf*)p;
      v8bf hi = *(const v8bf*)(p + 16);
      bfrag[ni] = __builtin_shufflevector(lo, hi, 0,1,2,3,4,5,6,7,8,9,10,11,12,13,14,15);
    }
#pragma unroll
    for (int mi = 0; mi < 2; ++mi)
#pragma unroll
      for (int ni = 0; ni < 4; ++ni)
        accK[mi][ni] = __builtin_amdgcn_wmma_f32_16x16x32_bf16(
            false, afrag[mi], false, bfrag[ni], (short)0, accK[mi][ni], false, false);
#pragma unroll
    for (int ni = 0; ni < 4; ++ni) {
      const __bf16* p = &lV[cur][(waveN * 64 + ni * 16 + lm) * LDS_STRIDE + 8 * g];
      v8bf lo = *(const v8bf*)p;
      v8bf hi = *(const v8bf*)(p + 16);
      bfrag[ni] = __builtin_shufflevector(lo, hi, 0,1,2,3,4,5,6,7,8,9,10,11,12,13,14,15);
    }
#pragma unroll
    for (int mi = 0; mi < 2; ++mi)
#pragma unroll
      for (int ni = 0; ni < 4; ++ni)
        accV[mi][ni] = __builtin_amdgcn_wmma_f32_16x16x32_bf16(
            false, afrag[mi], false, bfrag[ni], (short)0, accV[mi][ni], false, false);
  }

  // epilogue: bias + bf16 stores (C/D layout: lane col = lm, row m = j + 8g)
#pragma unroll
  for (int ni = 0; ni < 4; ++ni) {
    const int n = blockN + waveN * 64 + ni * 16 + lm;
    const float bvk = biask[n];
    const float bvv = biasv[n];
#pragma unroll
    for (int mi = 0; mi < 2; ++mi) {
#pragma unroll
      for (int j = 0; j < 8; ++j) {
        const int m = blockM + waveM * 32 + mi * 16 + j + 8 * g;
        OutK[(size_t)m * 1024 + n] = (__bf16)(accK[mi][ni][j] + bvk);
        OutV[(size_t)m * 1024 + n] = (__bf16)(accV[mi][ni][j] + bvv);
      }
    }
  }
}

// ---------------------------------------------------------------- kernel 4: row-0 attention (q0 = bq)
__global__ __launch_bounds__(256)
void attn_kernel(const __bf16* __restrict__ Kb, const __bf16* __restrict__ Vb,
                 const float* __restrict__ bq, const float* __restrict__ madd,
                 float* __restrict__ ctx0) {
  __shared__ float q[DH_];
  __shared__ float sc[S_];
  __shared__ float red[256];
  const int cb  = blockIdx.x / H_;
  const int h   = blockIdx.x % H_;
  const int tid = threadIdx.x;

  if (tid < DH_) q[tid] = bq[h * DH_ + tid];
  __syncthreads();

  for (int s = tid; s < S_; s += 256) {
    const __bf16* kp = Kb + (size_t)cb * S_ * D_ + (size_t)s * D_ + h * DH_;
    float a = 0.f;
#pragma unroll 16
    for (int d = 0; d < DH_; ++d) a += q[d] * (float)kp[d];
    sc[s] = a * 0.125f + madd[(size_t)cb * MAXT_ + s];
  }
  __syncthreads();

  float lmax = fmaxf(sc[tid], sc[tid + 256]);
  red[tid] = lmax; __syncthreads();
  for (int o = 128; o > 0; o >>= 1) { if (tid < o) red[tid] = fmaxf(red[tid], red[tid + o]); __syncthreads(); }
  const float mx = red[0];
  __syncthreads();
  const float e0 = __expf(sc[tid] - mx);
  const float e1 = __expf(sc[tid + 256] - mx);
  sc[tid] = e0; sc[tid + 256] = e1;
  red[tid] = e0 + e1; __syncthreads();
  for (int o = 128; o > 0; o >>= 1) { if (tid < o) red[tid] += red[tid + o]; __syncthreads(); }
  const float inv = 1.0f / red[0];
  __syncthreads();

  const int d   = tid & 63;
  const int grp = tid >> 6;
  float a = 0.f;
  const __bf16* vbase = Vb + (size_t)cb * S_ * D_ + h * DH_ + d;
#pragma unroll 4
  for (int s = grp * 128; s < grp * 128 + 128; ++s)
    a += sc[s] * (float)vbase[(size_t)s * D_];
  red[tid] = a; __syncthreads();
  if (tid < 64) {
    float r = (red[tid] + red[tid + 64] + red[tid + 128] + red[tid + 192]) * inv;
    ctx0[(size_t)cb * D_ + h * DH_ + tid] = r;
  }
}

// ---------------------------------------------------------------- kernel 5: small-M (80-row) GEMM, weights read once
__global__ __launch_bounds__(128)
void mlp_gemm_kernel(const float* __restrict__ X, const float* __restrict__ W,
                     const float* __restrict__ bias, float* __restrict__ Out,
                     int K, int Nn, int act) {
  __shared__ float lx[CB_ * 128];  // 40 KB
  const int tid = threadIdx.x;
  const int n = blockIdx.x * 128 + tid;

  float acc[CB_];
#pragma unroll
  for (int r = 0; r < CB_; ++r) acc[r] = 0.f;

  for (int k0 = 0; k0 < K; k0 += 128) {
    __syncthreads();
    for (int i = tid; i < CB_ * 128; i += 128) {
      const int r = i >> 7, k = i & 127;
      lx[i] = X[(size_t)r * K + k0 + k];
    }
    __syncthreads();
    for (int k = 0; k < 128; ++k) {
      const float w = W[(size_t)(k0 + k) * Nn + n];
#pragma unroll
      for (int r = 0; r < CB_; ++r) acc[r] += lx[r * 128 + k] * w;
    }
  }
  const float bv = bias[n];
  for (int r = 0; r < CB_; ++r) {
    float v = acc[r] + bv;
    if (act) v = 0.5f * v * (1.0f + erff(v * 0.70710678118654752f));  // exact gelu
    Out[(size_t)r * Nn + n] = v;
  }
}

// ---------------------------------------------------------------- kernel 6: LayerNorm over D; optional residual & pooled permute
__global__ __launch_bounds__(256)
void ln_kernel(const float* __restrict__ Xin, const float* __restrict__ Radd,
               const float* __restrict__ gg, const float* __restrict__ bb,
               float* __restrict__ Out, int toPooled) {
  __shared__ float red[256];
  const int cb = blockIdx.x;
  const int tid = threadIdx.x;
  float v[4];
  float s = 0.f;
#pragma unroll
  for (int i = 0; i < 4; ++i) {
    const int d = tid + i * 256;
    float x = Xin[(size_t)cb * D_ + d];
    if (Radd) x += Radd[(size_t)cb * D_ + d];
    v[i] = x; s += x;
  }
  red[tid] = s; __syncthreads();
  for (int o = 128; o > 0; o >>= 1) { if (tid < o) red[tid] += red[tid + o]; __syncthreads(); }
  const float mu = red[0] * (1.0f / D_);
  __syncthreads();
  float s2 = 0.f;
#pragma unroll
  for (int i = 0; i < 4; ++i) { const float d0 = v[i] - mu; s2 += d0 * d0; }
  red[tid] = s2; __syncthreads();
  for (int o = 128; o > 0; o >>= 1) { if (tid < o) red[tid] += red[tid + o]; __syncthreads(); }
  const float var = red[0] * (1.0f / D_);
  const float inv = rsqrtf(var + LN_EPS);
#pragma unroll
  for (int i = 0; i < 4; ++i) {
    const int d = tid + i * 256;
    const float y = (v[i] - mu) * inv * gg[d] + bb[d];
    if (toPooled) {
      const int c = cb / B_, b = cb % B_;
      Out[((size_t)b * C_ + c) * D_ + d] = y;
    } else {
      Out[(size_t)cb * D_ + d] = y;
    }
  }
}

// ---------------------------------------------------------------- launch
extern "C" void kernel_launch(void* const* d_in, const int* in_sizes, int n_in,
                              void* d_out, int out_size, void* d_ws, size_t ws_size,
                              hipStream_t stream) {
  (void)in_sizes; (void)n_in; (void)out_size; (void)ws_size;
  const float* vision = (const float*)d_in[0];
  const int*   cstate = (const int*)d_in[1];
  // d_in[2] = Wq (unused: q row 0 is exactly bq since x row 0 is the zero prefix)
  const float* bq  = (const float*)d_in[3];
  const float* Wk  = (const float*)d_in[4];
  const float* bk  = (const float*)d_in[5];
  const float* Wv  = (const float*)d_in[6];
  const float* bv  = (const float*)d_in[7];
  const float* Wo  = (const float*)d_in[8];
  const float* bo  = (const float*)d_in[9];
  const float* g1  = (const float*)d_in[10];
  const float* b1  = (const float*)d_in[11];
  const float* Wi  = (const float*)d_in[12];
  const float* bi  = (const float*)d_in[13];
  const float* Wo2 = (const float*)d_in[14];
  const float* bo2 = (const float*)d_in[15];
  const float* g2  = (const float*)d_in[16];
  const float* b2  = (const float*)d_in[17];

  float* out = (float*)d_out;
  float* cluster_out = out;
  float* mask_out    = out + OFF_MASK;
  float* pooled_out  = out + OFF_POOL;

  // workspace carve-up
  char* ws = (char*)d_ws;
  size_t off = 0;
  auto carve = [&](size_t bytes) { char* p = ws + off; off = (off + bytes + 255) & ~(size_t)255; return p; };
  __bf16* Xb   = (__bf16*)carve((size_t)M_ * D_ * 2);
  __bf16* Wkt  = (__bf16*)carve((size_t)D_ * D_ * 2);
  __bf16* Wvt  = (__bf16*)carve((size_t)D_ * D_ * 2);
  __bf16* Kb   = (__bf16*)carve((size_t)M_ * D_ * 2);
  __bf16* Vb   = (__bf16*)carve((size_t)M_ * D_ * 2);
  float*  ctx0 = (float*)carve((size_t)CB_ * D_ * 4);
  float*  R    = (float*)carve((size_t)CB_ * D_ * 4);
  float*  H0   = (float*)carve((size_t)CB_ * D_ * 4);
  float*  I    = (float*)carve((size_t)CB_ * DFF_ * 4);
  float*  O    = (float*)carve((size_t)CB_ * D_ * 4);

  // 1) weight transpose/convert (bf16 [N][K])
  wconv_kernel<<<dim3((D_ * D_) / 256), dim3(256), 0, stream>>>(Wk, Wkt);
  wconv_kernel<<<dim3((D_ * D_) / 256), dim3(256), 0, stream>>>(Wv, Wvt);

  // 2) gather: cluster, cluster_mask, X(bf16)
  gather_kernel<<<dim3(CB_), dim3(256), 0, stream>>>(vision, cstate, cluster_out, mask_out, Xb);

  // 3) fused K|V = X@{Wk,Wv} + {bk,bv}  (WMMA bf16, async double-buffered LDS)
  gemm_kv_fused<<<dim3(1024 / 128, M_ / 128), dim3(256), 0, stream>>>(
      Xb, Wkt, Wvt, bk, bv, Kb, Vb);

  // 4) attention for query row 0 only (q0 = bq); madd = cluster_mask[:, :S]
  attn_kernel<<<dim3(CB_ * H_), dim3(256), 0, stream>>>(Kb, Vb, bq, mask_out, ctx0);

  // 5) row-0 MLP chain
  mlp_gemm_kernel<<<dim3(D_ / 128),   dim3(128), 0, stream>>>(ctx0, Wo,  bo,  R, D_,   D_,   0);
  ln_kernel<<<dim3(CB_), dim3(256), 0, stream>>>(R, nullptr, g1, b1, H0, 0);
  mlp_gemm_kernel<<<dim3(DFF_ / 128), dim3(128), 0, stream>>>(H0,  Wi,  bi,  I, D_,   DFF_, 1);
  mlp_gemm_kernel<<<dim3(D_ / 128),   dim3(128), 0, stream>>>(I,   Wo2, bo2, O, DFF_, D_,   0);
  ln_kernel<<<dim3(CB_), dim3(256), 0, stream>>>(O, H0, g2, b2, pooled_out, 1);
}
